// LSTMAttention_17231408791857
// MI455X (gfx1250) — matrix operand order repaired and checked by reading, not compiled, not run on hardware
//
#include <hip/hip_runtime.h>
#include <hip/hip_bf16.h>

// ---------------- problem constants ----------------
#define BB   32      // batch
#define TT   2048    // time steps
#define FF   512     // features
#define HH   512     // hidden
#define NWG  32      // persistent workgroups (grid barrier participants)
#define SLICE 16     // H columns owned per WG
#define THREADS 256  // 8 wave32s

typedef __bf16 bf16;
typedef __attribute__((ext_vector_type(16))) __bf16 v16bf;
typedef __attribute__((ext_vector_type(8)))  __bf16 v8bf;
typedef __attribute__((ext_vector_type(4)))  __bf16 v4bf;
typedef __attribute__((ext_vector_type(8)))  float  v8f;
typedef __attribute__((ext_vector_type(4)))  float  v4f;

// LDS row strides (elements), padded so 16 consecutive rows hit distinct banks
#define WG_STR 1032   // gate weights rows of 1024
#define WS_STR 520    // W_hs/W_cs rows of 512
#define A_STR  1032   // activation rows of 1024
#define G_STR  64     // gates scratch [32][64] f32

struct SharedMem {
  bf16  Wg[64 * WG_STR];          // [gatecol 0..63][k 0..1023]  (i,f,g,o blocks of 16)
  bf16  Wsc[2][SLICE * WS_STR];   // [0]=W_hs slice, [1]=W_cs slice
  bf16  Abuf[BB * A_STR];         // staging: [xi|h] then [h_new|c_new]
  float gates[BB * G_STR];        // GEMM1 result
  float out2[2][2][BB][SLICE];    // [khalf][h2/c2] partial results
  float c_loc[BB][SLICE];         // local cell-state carry (never leaves WGP)
  float bias[64];                 // b_ih + b_hh for our 64 gate columns
};

__device__ __forceinline__ float sigmoidf(float x) { return 1.0f / (1.0f + expf(-x)); }

// CDNA5 async copy: 16B global -> LDS per lane, tracked by ASYNCcnt.
__device__ __forceinline__ void async_ld16(unsigned lds_off, unsigned long long gaddr) {
  asm volatile("global_load_async_to_lds_b128 %0, %1, off"
               :: "v"(lds_off), "v"(gaddr) : "memory");
}
__device__ __forceinline__ void wait_async0() {
  asm volatile("s_wait_asynccnt 0" ::: "memory");
}

// A fragment: 16x32 bf16 tile, row-major in LDS.
// lanes 0-15: row=lane, elems 0..7 = K 0..7, elems 8..15 = K 16..23
// lanes 16-31: row=lane-16, K 8..15 and 24..31
__device__ __forceinline__ v16bf ld_a(const bf16* base, int stride, int lane) {
  int half = (lane >> 4) & 1, r = lane & 15;
  const bf16* p = base + r * stride + half * 8;
  v8bf lo = *(const v8bf*)p;
  v8bf hi = *(const v8bf*)(p + 16);
  v16bf o;
#pragma unroll
  for (int i = 0; i < 8; ++i) { o[i] = lo[i]; o[i + 8] = hi[i]; }
  return o;
}

// B fragment: 32x16 (KxN) from weights stored row-major W[n][k].
// lanes 0-15: col=lane, K 0..7 and 16..23 ; lanes 16-31: col=lane-16, K 8..15 and 24..31
__device__ __forceinline__ v16bf ld_b(const bf16* wrow0, int stride, int lane) {
  int half = (lane >> 4) & 1, n = lane & 15;
  const bf16* p = wrow0 + n * stride + half * 8;
  v8bf lo = *(const v8bf*)p;
  v8bf hi = *(const v8bf*)(p + 16);
  v16bf o;
#pragma unroll
  for (int i = 0; i < 8; ++i) { o[i] = lo[i]; o[i + 8] = hi[i]; }
  return o;
}

// Monotonic-generation grid barrier over NWG workgroups.
__device__ __forceinline__ void grid_barrier(unsigned* bar, unsigned gen) {
  __threadfence();        // release: flush our global stores to device scope
  __syncthreads();
  if (threadIdx.x == 0) {
    atomicAdd(bar, 1u);
    unsigned target = gen * NWG;
    while (__hip_atomic_load(bar, __ATOMIC_ACQUIRE, __HIP_MEMORY_SCOPE_AGENT) < target)
      __builtin_amdgcn_s_sleep(1);
  }
  __syncthreads();
  __threadfence();        // acquire: invalidate WGP caches before re-reading
}

__global__ void __launch_bounds__(THREADS, 1)
lstm_attn_persistent(const float* __restrict__ X,
                     const float* __restrict__ W_ih, const float* __restrict__ W_hh,
                     const float* __restrict__ b_ih, const float* __restrict__ b_hh,
                     const float* __restrict__ W_hs, const float* __restrict__ W_cs,
                     float* __restrict__ out,
                     bf16* __restrict__ h_buf, bf16* __restrict__ hn_buf,
                     bf16* __restrict__ cn_buf, unsigned* __restrict__ bar)
{
  __shared__ SharedMem sm;
  const int tid  = threadIdx.x;
  const int wg   = blockIdx.x;       // 0..31
  const int lane = tid & 31;
  const int wave = tid >> 5;
  const int col0 = wg * SLICE;       // our H-slice base

  // ---------- one-time: stage weights (f32 -> bf16) into LDS ----------
  for (int i = tid; i < 64 * 1024; i += THREADS) {
    int n = i >> 10, k = i & 1023;
    int j = (n >> 4) * HH + col0 + (n & 15);           // global 4H row (i,f,g,o order)
    float wv = (k < 512) ? W_ih[(size_t)j * FF + k] : W_hh[(size_t)j * HH + (k - 512)];
    sm.Wg[n * WG_STR + k] = (bf16)wv;
  }
  for (int i = tid; i < SLICE * 512; i += THREADS) {
    int c = i >> 9, k = i & 511;
    sm.Wsc[0][c * WS_STR + k] = (bf16)W_hs[(size_t)(col0 + c) * FF + k];
    sm.Wsc[1][c * WS_STR + k] = (bf16)W_cs[(size_t)(col0 + c) * FF + k];
  }
  if (tid < 64) {
    int j = (tid >> 4) * HH + col0 + (tid & 15);
    sm.bias[tid] = b_ih[j] + b_hh[j];
  }
  // zero initial carries (h in global, c local)
  for (int i = tid; i < BB * SLICE; i += THREADS) {
    int b = i >> 4, c = i & 15;
    sm.c_loc[b][c] = 0.0f;
    h_buf[b * HH + col0 + c] = (bf16)0.0f;
  }
  unsigned gen = 0;
  grid_barrier(bar, ++gen);

  // ---------- sequential scan over time ----------
  for (int t = 0; t < TT; ++t) {
    // async-stage h (bf16, L2) into Abuf[:,512:1024] while we convert X on VALU
    for (int v = tid; v < BB * (HH / 8); v += THREADS) {
      int b = v >> 6, kc = (v & 63) << 3;
      unsigned lds = (unsigned)(uintptr_t)&sm.Abuf[b * A_STR + 512 + kc];
      async_ld16(lds, (unsigned long long)(uintptr_t)(h_buf + b * HH + kc));
    }
    // xi: f32 -> bf16 into Abuf[:,0:512]
    for (int v = tid; v < BB * (FF / 4); v += THREADS) {
      int b = v >> 7, kc = (v & 127) << 2;
      v4f x = *(const v4f*)(X + ((size_t)b * TT + t) * FF + kc);
      v4bf xb; xb[0] = (bf16)x[0]; xb[1] = (bf16)x[1]; xb[2] = (bf16)x[2]; xb[3] = (bf16)x[3];
      *(v4bf*)&sm.Abuf[b * A_STR + kc] = xb;
    }
    // warm L2/L0 with next timestep's X while the GEMMs run
    if (t + 1 < TT) {
      for (int i = tid; i < BB * 16; i += THREADS) {
        int b = i >> 4, ln = i & 15;
        __builtin_prefetch(X + ((size_t)b * TT + (t + 1)) * FF + ln * 32, 0, 1);
      }
    }
    wait_async0();
    __syncthreads();

    // GEMM1: gates[32x64] = [xi|h] (32x1024) @ Wg^T ; 8 waves = 2 Mtiles x 4 gate tiles
    // two interleaved accumulators to break the WMMA D->C dependence chain
    {
      int mt = wave & 1, nt = wave >> 1;
      v8f acc0 = {}, acc1 = {};
#pragma unroll 8
      for (int ks = 0; ks < 32; ks += 2) {
        v16bf a0 = ld_a(&sm.Abuf[(mt * 16) * A_STR + ks * 32], A_STR, lane);
        v16bf b0 = ld_b(&sm.Wg[(nt * 16) * WG_STR + ks * 32], WG_STR, lane);
        acc0 = __builtin_amdgcn_wmma_f32_16x16x32_bf16(false, a0, false, b0,
                                                       (short)0, acc0, false, false);
        v16bf a1 = ld_a(&sm.Abuf[(mt * 16) * A_STR + (ks + 1) * 32], A_STR, lane);
        v16bf b1 = ld_b(&sm.Wg[(nt * 16) * WG_STR + (ks + 1) * 32], WG_STR, lane);
        acc1 = __builtin_amdgcn_wmma_f32_16x16x32_bf16(false, a1, false, b1,
                                                       (short)0, acc1, false, false);
      }
      int hf = lane >> 4, nn = lane & 15;
#pragma unroll
      for (int r = 0; r < 8; ++r)
        sm.gates[(mt * 16 + hf * 8 + r) * G_STR + nt * 16 + nn] = acc0[r] + acc1[r];
    }
    __syncthreads();

    // elementwise LSTM cell on our 16-column slice
    for (int i = tid; i < BB * SLICE; i += THREADS) {
      int b = i >> 4, ml = i & 15;
      float ig = sm.gates[b * G_STR + ml]      + sm.bias[ml];
      float fg = sm.gates[b * G_STR + 16 + ml] + sm.bias[16 + ml];
      float gg = sm.gates[b * G_STR + 32 + ml] + sm.bias[32 + ml];
      float og = sm.gates[b * G_STR + 48 + ml] + sm.bias[48 + ml];
      float cn = sigmoidf(fg) * sm.c_loc[b][ml] + sigmoidf(ig) * tanhf(gg);
      float hn = sigmoidf(og) * tanhf(cn);
      hn_buf[b * HH + col0 + ml] = (bf16)hn;
      cn_buf[b * HH + col0 + ml] = (bf16)cn;
    }
    grid_barrier(bar, ++gen);

    // async-stage Abuf = [ h_new | c_new ] (both bf16 from L2)
    for (int v = tid; v < BB * (HH / 8); v += THREADS) {
      int b = v >> 6, kc = (v & 63) << 3;
      unsigned lds0 = (unsigned)(uintptr_t)&sm.Abuf[b * A_STR + kc];
      unsigned lds1 = (unsigned)(uintptr_t)&sm.Abuf[b * A_STR + 512 + kc];
      async_ld16(lds0, (unsigned long long)(uintptr_t)(hn_buf + b * HH + kc));
      async_ld16(lds1, (unsigned long long)(uintptr_t)(cn_buf + b * HH + kc));
    }
    wait_async0();
    __syncthreads();

    // GEMM2: h2 = h_new @ W_hs_slice^T ; c2 = c_new @ W_cs_slice^T
    // 8 wave-tasks: (sel in {h2,c2}) x (Mtile) x (K half); partials reduced below
    {
      int mt = wave & 1, sel = (wave >> 1) & 1, kh = wave >> 2;
      v8f acc = {};
#pragma unroll
      for (int ks = 0; ks < 8; ++ks) {
        v16bf a  = ld_a(&sm.Abuf[(mt * 16) * A_STR + sel * 512 + kh * 256 + ks * 32],
                        A_STR, lane);
        v16bf bb = ld_b(&sm.Wsc[sel][kh * 256 + ks * 32], WS_STR, lane);
        acc = __builtin_amdgcn_wmma_f32_16x16x32_bf16(false, a, false, bb,
                                                      (short)0, acc, false, false);
      }
      int hf = lane >> 4, nn = lane & 15;
#pragma unroll
      for (int r = 0; r < 8; ++r)
        sm.out2[kh][sel][mt * 16 + hf * 8 + r][nn] = acc[r];
    }
    __syncthreads();

    // fused output + carry update:
    // att = tanh(h2+c2); o = tanh(att + x); out = x + tanh(o); carry = (h2, c2)
    for (int i = tid; i < BB * SLICE; i += THREADS) {
      int b = i >> 4, ml = i & 15;
      float h2 = sm.out2[0][0][b][ml] + sm.out2[1][0][b][ml];
      float c2 = sm.out2[0][1][b][ml] + sm.out2[1][1][b][ml];
      size_t xo = ((size_t)b * TT + t) * FF + col0 + ml;
      float x = X[xo];
      float att = tanhf(h2 + c2);
      float o = tanhf(att + x);
      out[xo] = x + tanhf(o);
      sm.c_loc[b][ml] = c2;
      h_buf[b * HH + col0 + ml] = (bf16)h2;
    }
    grid_barrier(bar, ++gen);
  }
}

extern "C" void kernel_launch(void* const* d_in, const int* in_sizes, int n_in,
                              void* d_out, int out_size, void* d_ws, size_t ws_size,
                              hipStream_t stream) {
  const float* X    = (const float*)d_in[0];
  const float* W_ih = (const float*)d_in[1];
  const float* W_hh = (const float*)d_in[2];
  const float* b_ih = (const float*)d_in[3];
  const float* b_hh = (const float*)d_in[4];
  const float* W_hs = (const float*)d_in[5];
  const float* W_cs = (const float*)d_in[6];
  float* out = (float*)d_out;

  char* ws = (char*)d_ws;
  unsigned* bar = (unsigned*)ws;                 // 256 B barrier region
  bf16* h_buf  = (bf16*)(ws + 256);              // 32*512 bf16 = 32 KB
  bf16* hn_buf = (bf16*)(ws + 256 + 32768);      // 32 KB
  bf16* cn_buf = (bf16*)(ws + 256 + 65536);      // 32 KB

  hipMemsetAsync(bar, 0, 256, stream);           // graph-capture-safe
  lstm_attn_persistent<<<NWG, THREADS, 0, stream>>>(
      X, W_ih, W_hh, b_ih, b_hh, W_hs, W_cs, out, h_buf, hn_buf, cn_buf, bar);
}
